// DeepSeekMoE_7267084664932
// MI455X (gfx1250) — compile-verified, matrix operand order checked by reading, not run on hardware
//
#include <hip/hip_runtime.h>
#include <stdint.h>

#define T_TOK 2048
#define DDIM  1024
#define HDIM  704
#define NEXP  16
#define KACT  4
#define HSH   1408
#define ATOT  (T_TOK*KACT)
#define ROUTE_SCALE 2.5446f

typedef __attribute__((ext_vector_type(16))) __bf16 v16bf;
typedef __attribute__((ext_vector_type(2)))  __bf16 bf16x2;
typedef __attribute__((ext_vector_type(2)))  float  f32x2;
typedef __attribute__((ext_vector_type(8)))  float  v8f;

// ---------------- helpers ----------------

__device__ __forceinline__ uint16_t f2bf(float f) {
    union { __bf16 h; uint16_t u; } v;
    v.h = (__bf16)f;
    return v.u;
}

// packed fp32x2 -> bf16x2 via vector fptrunc (lowers to v_cvt_pk_bf16_f32)
__device__ __forceinline__ uint32_t pkbf(float lo, float hi) {
    f32x2 f; f.x = lo; f.y = hi;
    union { bf16x2 h; uint32_t u; } v;
    v.h = __builtin_convertvector(f, bf16x2);
    return v.u;
}

// generic->LDS byte address (low 32 bits of the shared-aperture address)
__device__ __forceinline__ uint32_t lds_off(const void* p) {
    return (uint32_t)(uintptr_t)p;
}

// CDNA5 async copy global -> LDS, 16B per lane, tracked by ASYNCcnt
__device__ __forceinline__ void async_g2l_b128(uint32_t ldsaddr, const void* g) {
    asm volatile("global_load_async_to_lds_b128 %0, %1, off"
                 :: "v"(ldsaddr), "v"((uint64_t)(uintptr_t)g)
                 : "memory");
}
__device__ __forceinline__ void wait_async_le2() {
    asm volatile("s_wait_asynccnt 0x2" ::: "memory");
}
__device__ __forceinline__ void wait_async_0() {
    asm volatile("s_wait_asynccnt 0x0" ::: "memory");
}

// Load one 16x32 bf16 fragment from an LDS tile (row-major, stride 32).
// CDNA5 ISA 7.12.2 layout: lanes 0-15 rows 0-15 K=0..7/16..23, lanes 16-31
// same rows K=8..15/24..31.
__device__ __forceinline__ v16bf lds_frag(const uint16_t* base, int lane) {
    int r  = lane & 15;
    int kh = (lane >> 4) << 3;
    const uint16_t* p = base + r * 32 + kh;
    union { uint4 q[2]; v16bf v; } u;
    u.q[0] = *(const uint4*)(p);
    u.q[1] = *(const uint4*)(p + 16);
    return u.v;
}

__device__ __forceinline__ v8f wmma_bf16(v16bf a, v16bf b, v8f c) {
    return __builtin_amdgcn_wmma_f32_16x16x32_bf16(
        false, a, false, b, (short)0, c, false, false);
}

// ---------------- kernel 1: x fp32 -> bf16 ----------------

__global__ __launch_bounds__(256)
void convert_x_kernel(const float* __restrict__ x, uint16_t* __restrict__ xb) {
    size_t i = ((size_t)blockIdx.x * 256 + threadIdx.x) * 4;
    if (i >= (size_t)T_TOK * DDIM) return;
    float4 v = *(const float4*)(x + i);
    uint2 o;
    o.x = pkbf(v.x, v.y);
    o.y = pkbf(v.z, v.w);
    *(uint2*)(xb + i) = o;
}

// ---------------- kernel 2: sigmoid gate + group-limited top-k ----------------

__global__ __launch_bounds__(128)
void gate_kernel(const float* __restrict__ x, const float* __restrict__ gw,
                 const float* __restrict__ gb,
                 int* __restrict__ widx, float* __restrict__ wval) {
    int wave = threadIdx.x >> 5, lane = threadIdx.x & 31;
    int t = blockIdx.x * 4 + wave;
    if (t >= T_TOK) return;

    float acc[NEXP];
#pragma unroll
    for (int e = 0; e < NEXP; e++) acc[e] = 0.f;
    const float* xr = x + (size_t)t * DDIM;
    for (int i = lane; i < DDIM; i += 32) {
        float xv = xr[i];
#pragma unroll
        for (int e = 0; e < NEXP; e++) acc[e] += xv * gw[e * DDIM + i];
    }
#pragma unroll
    for (int e = 0; e < NEXP; e++)
        for (int off = 16; off > 0; off >>= 1)
            acc[e] += __shfl_xor(acc[e], off, 32);

    if (lane == 0) {
        float s[NEXP], rt[NEXP];
        for (int e = 0; e < NEXP; e++) {
            s[e]  = 1.f / (1.f + __expf(-acc[e]));
            rt[e] = s[e] + gb[e];
        }
        float gs[4];
        for (int g = 0; g < 4; g++) {
            float m1 = -3e38f, m2 = -3e38f;
            for (int j = 0; j < 4; j++) {
                float v = rt[g * 4 + j];
                if (v > m1) { m2 = m1; m1 = v; } else if (v > m2) m2 = v;
            }
            gs[g] = m1 + m2;
        }
        int g1 = 0;
        for (int g = 1; g < 4; g++) if (gs[g] > gs[g1]) g1 = g;
        int g2 = (g1 == 0) ? 1 : 0;
        for (int g = 0; g < 4; g++) if (g != g1 && gs[g] > gs[g2]) g2 = g;
        for (int e = 0; e < NEXP; e++) {
            int g = e >> 2;
            if (g != g1 && g != g2) rt[e] = -3e38f;
        }
        int id[KACT]; float wv[KACT]; float sum = 0.f;
        for (int j = 0; j < KACT; j++) {
            int b = 0; float bv = -3.5e38f;
            for (int e = 0; e < NEXP; e++) if (rt[e] > bv) { bv = rt[e]; b = e; }
            id[j] = b; rt[b] = -3.5e38f;
            wv[j] = s[b]; sum += wv[j];
        }
        float inv = ROUTE_SCALE / fmaxf(sum, 1e-9f);
        for (int j = 0; j < KACT; j++) {
            widx[t * KACT + j] = id[j];
            wval[t * KACT + j] = wv[j] * inv;
        }
    }
}

// ---------------- kernel 3: deterministic counting-sort dispatch ----------------

__global__ __launch_bounds__(256)
void dispatch_kernel(const int* __restrict__ widx, int* __restrict__ perm,
                     int* __restrict__ offsets) {
    __shared__ int cnt[NEXP * 256];
    int tid = threadIdx.x;
    const int CH = ATOT / 256;
    int c[NEXP];
#pragma unroll
    for (int e = 0; e < NEXP; e++) c[e] = 0;
    int base = tid * CH;
    for (int j = 0; j < CH; j++) c[widx[base + j]]++;
    for (int e = 0; e < NEXP; e++) cnt[e * 256 + tid] = c[e];
    __syncthreads();
    if (tid == 0) {
        int run = 0;
        for (int e = 0; e < NEXP; e++) {
            offsets[e] = run;
            for (int i = 0; i < 256; i++) {
                int v = cnt[e * 256 + i];
                cnt[e * 256 + i] = run;
                run += v;
            }
        }
        offsets[NEXP] = run;
    }
    __syncthreads();
    int pos[NEXP];
    for (int e = 0; e < NEXP; e++) pos[e] = cnt[e * 256 + tid];
    for (int j = 0; j < CH; j++) {
        int a = base + j;
        int e = widx[a];
        perm[pos[e]++] = a;
    }
}

// ---------------- kernel 4/5: up+gate proj with fused SwiGLU ----------------
// M=128 x N=64 tile, K-step 32.
//  - A-tile: async global->LDS (gathered per-lane addresses), double-buffered,
//    issued one step ahead; per-wave s_wait_asynccnt placed right after the
//    barrier while loadcnt is already drained (no forced serialization).
//  - B-tiles: fp32 weights register-prefetched; next step's loads issued AFTER
//    the asynccnt wait so they overlap fragment loads + all 8 WMMAs.
//  - B fragments software-pipelined so WMMAs issue back-to-back.

__global__ __launch_bounds__(256)
void pass1_swiglu(const uint16_t* __restrict__ xb,
                  const float* __restrict__ W1base,
                  const float* __restrict__ W3base,
                  uint16_t* __restrict__ act,
                  const int* __restrict__ perm,
                  const int* __restrict__ offsets,
                  int mode, int Ntot, int Ktot, int actLd) {
    int e = blockIdx.z;
    const float* W1 = W1base + (size_t)e * Ntot * Ktot;
    const float* W3 = W3base + (size_t)e * Ntot * Ktot;
    int h0 = blockIdx.x * 64;
    int rowStart, rowEnd;
    if (mode == 0) {
        rowStart = offsets[e] + blockIdx.y * 128;
        rowEnd   = offsets[e + 1];
        if (rowStart >= rowEnd) return;
    } else {
        rowStart = blockIdx.y * 128;
        rowEnd   = T_TOK;
    }

    __shared__ __align__(16) uint16_t sA[2][128 * 32];
    __shared__ __align__(16) uint16_t sB1[64 * 32];
    __shared__ __align__(16) uint16_t sB3[64 * 32];
    __shared__ int toks[128];

    int tid = threadIdx.x;
    if (tid < 128) {
        int a = rowStart + tid;
        toks[tid] = (a < rowEnd) ? ((mode == 0) ? (perm[a] >> 2) : a) : 0;
    }
    __syncthreads();

    int wave = tid >> 5, lane = tid & 31;

    int ar = tid >> 1, ac = (tid & 1) << 4;
    const uint16_t* aSrc = xb + (size_t)toks[ar] * Ktot + ac;
    uint32_t aDst0 = lds_off(&sA[0][ar * 32 + ac]);
    uint32_t aDst1 = lds_off(&sA[1][ar * 32 + ac]);

    int br = tid >> 2, bc = (tid & 3) << 3;
    const float* pW1 = W1 + (size_t)(h0 + br) * Ktot + bc;
    const float* pW3 = W3 + (size_t)(h0 + br) * Ktot + bc;

    // prologue: async A(0) into buffer 0, B(0) into registers
    async_g2l_b128(aDst0,      aSrc);
    async_g2l_b128(aDst0 + 16, aSrc + 8);
    float4 r1a = *(const float4*)(pW1);
    float4 r1b = *(const float4*)(pW1 + 4);
    float4 r3a = *(const float4*)(pW3);
    float4 r3b = *(const float4*)(pW3 + 4);

    v8f accG[4], accU[4];
#pragma unroll
    for (int n = 0; n < 4; n++)
#pragma unroll
        for (int i = 0; i < 8; i++) { accG[n][i] = 0.f; accU[n][i] = 0.f; }

    int p = 0;
    for (int k0 = 0; k0 < Ktot; k0 += 32, p ^= 1) {
        bool more = (k0 + 32) < Ktot;
        __syncthreads();   // previous compute finished: sB / sA[p^1] reusable
        {   // store B(k) registers -> LDS (packed hw convert)
            uint32_t* d1 = (uint32_t*)&sB1[br * 32 + bc];
            uint32_t* d3 = (uint32_t*)&sB3[br * 32 + bc];
            d1[0] = pkbf(r1a.x, r1a.y); d1[1] = pkbf(r1a.z, r1a.w);
            d1[2] = pkbf(r1b.x, r1b.y); d1[3] = pkbf(r1b.z, r1b.w);
            d3[0] = pkbf(r3a.x, r3a.y); d3[1] = pkbf(r3a.z, r3a.w);
            d3[2] = pkbf(r3b.x, r3b.y); d3[3] = pkbf(r3b.z, r3b.w);
        }
        if (more) {   // async A(k+1) into the other buffer
            const uint16_t* s = aSrc + k0 + 32;
            uint32_t dst = (p == 0) ? aDst1 : aDst0;
            async_g2l_b128(dst,      s);
            async_g2l_b128(dst + 16, s + 8);
        }
        __syncthreads();   // B(k) visible block-wide

        // A(k) landed (per-wave rows); loadcnt already 0 here so the
        // conservative pre-asm drain is free.
        if (more) wait_async_le2(); else wait_async_0();

        if (more) {
            // B(k+1) global loads: overlap fragment loads + all 8 WMMAs
            r1a = *(const float4*)(pW1 + k0 + 32);
            r1b = *(const float4*)(pW1 + k0 + 36);
            r3a = *(const float4*)(pW3 + k0 + 32);
            r3b = *(const float4*)(pW3 + k0 + 36);
            if (k0 + 64 < Ktot) {
                __builtin_prefetch(pW1 + k0 + 64, 0, 0);
                __builtin_prefetch(pW3 + k0 + 64, 0, 0);
            }
        }

        v16bf aF = lds_frag(&sA[p][(wave * 16) * 32], lane);
        v16bf c1 = lds_frag(&sB1[0], lane);
        v16bf c3 = lds_frag(&sB3[0], lane);
#pragma unroll
        for (int nt = 0; nt < 4; nt++) {
            v16bf n1 = c1, n3 = c3;
            if (nt < 3) {   // pipeline: fetch next fragments before issuing WMMAs
                n1 = lds_frag(&sB1[(nt + 1) * 16 * 32], lane);
                n3 = lds_frag(&sB3[(nt + 1) * 16 * 32], lane);
            }
            accG[nt] = wmma_bf16(aF, c1, accG[nt]);
            accU[nt] = wmma_bf16(aF, c3, accU[nt]);
            c1 = n1; c3 = n3;
        }
    }
    // fused SwiGLU epilogue.  C layout: (m = i + 8*(lane>>4), n = lane&15)
#pragma unroll
    for (int nt = 0; nt < 4; nt++) {
#pragma unroll
        for (int i = 0; i < 8; i++) {
            int m = i + ((lane >> 4) << 3);
            int a = rowStart + wave * 16 + m;
            if (a < rowEnd) {
                float g = accG[nt][i], u = accU[nt][i];
                float sv = g / (1.f + __expf(-g));
                int n = h0 + nt * 16 + (lane & 15);
                act[(size_t)a * actLd + n] = f2bf(sv * u);
            }
        }
    }
}

// ---------------- kernel 6: down projection ----------------

__global__ __launch_bounds__(256)
void pass2_down(const uint16_t* __restrict__ act,
                const float* __restrict__ W2base,
                float* __restrict__ outBuf,
                const int* __restrict__ perm,
                const int* __restrict__ offsets,
                const float* __restrict__ wval,
                int mode, int Ktot, int actLd) {
    int e = blockIdx.z;
    const float* W2 = W2base + (size_t)e * DDIM * Ktot;
    int d0 = blockIdx.x * 64;
    int rowStart, rowEnd;
    if (mode == 0) {
        rowStart = offsets[e] + blockIdx.y * 128;
        rowEnd   = offsets[e + 1];
        if (rowStart >= rowEnd) return;
    } else {
        rowStart = blockIdx.y * 128;
        rowEnd   = T_TOK;
    }

    __shared__ __align__(16) uint16_t sA[2][128 * 32];
    __shared__ __align__(16) uint16_t sB[64 * 32];

    int tid = threadIdx.x, wave = tid >> 5, lane = tid & 31;

    int ar = tid >> 1, ac = (tid & 1) << 4;
    int arow = rowStart + ar;
    if (arow >= rowEnd) arow = rowEnd - 1;
    const uint16_t* aSrc = act + (size_t)arow * actLd + ac;
    uint32_t aDst0 = lds_off(&sA[0][ar * 32 + ac]);
    uint32_t aDst1 = lds_off(&sA[1][ar * 32 + ac]);

    int br = tid >> 2, bc = (tid & 3) << 3;
    const float* pW2 = W2 + (size_t)(d0 + br) * Ktot + bc;

    async_g2l_b128(aDst0,      aSrc);
    async_g2l_b128(aDst0 + 16, aSrc + 8);
    float4 r2a = *(const float4*)(pW2);
    float4 r2b = *(const float4*)(pW2 + 4);

    v8f acc[4];
#pragma unroll
    for (int n = 0; n < 4; n++)
#pragma unroll
        for (int i = 0; i < 8; i++) acc[n][i] = 0.f;

    int p = 0;
    for (int k0 = 0; k0 < Ktot; k0 += 32, p ^= 1) {
        bool more = (k0 + 32) < Ktot;
        __syncthreads();
        {
            uint32_t* dd = (uint32_t*)&sB[br * 32 + bc];
            dd[0] = pkbf(r2a.x, r2a.y); dd[1] = pkbf(r2a.z, r2a.w);
            dd[2] = pkbf(r2b.x, r2b.y); dd[3] = pkbf(r2b.z, r2b.w);
        }
        if (more) {
            const uint16_t* s = aSrc + k0 + 32;
            uint32_t dst = (p == 0) ? aDst1 : aDst0;
            async_g2l_b128(dst,      s);
            async_g2l_b128(dst + 16, s + 8);
        }
        __syncthreads();

        if (more) wait_async_le2(); else wait_async_0();

        if (more) {
            r2a = *(const float4*)(pW2 + k0 + 32);
            r2b = *(const float4*)(pW2 + k0 + 36);
            if (k0 + 64 < Ktot) __builtin_prefetch(pW2 + k0 + 64, 0, 0);
        }

        v16bf aF = lds_frag(&sA[p][(wave * 16) * 32], lane);
        v16bf cB = lds_frag(&sB[0], lane);
#pragma unroll
        for (int nt = 0; nt < 4; nt++) {
            v16bf nB = cB;
            if (nt < 3) nB = lds_frag(&sB[(nt + 1) * 16 * 32], lane);
            acc[nt] = wmma_bf16(aF, cB, acc[nt]);
            cB = nB;
        }
    }
#pragma unroll
    for (int nt = 0; nt < 4; nt++) {
#pragma unroll
        for (int i = 0; i < 8; i++) {
            int m = i + ((lane >> 4) << 3);
            int a = rowStart + wave * 16 + m;
            if (a < rowEnd) {
                int n = d0 + nt * 16 + (lane & 15);
                float v = acc[nt][i];
                if (mode == 0) {
                    int pk = perm[a];
                    outBuf[(size_t)pk * DDIM + n] = v * wval[pk];
                } else {
                    outBuf[(size_t)a * DDIM + n] = v;
                }
            }
        }
    }
}

// ---------------- kernel 7: deterministic combine ----------------

__global__ __launch_bounds__(256)
void combine_kernel(float* __restrict__ out, const float* __restrict__ part) {
    int i = blockIdx.x * 256 + threadIdx.x;
    if (i >= T_TOK * DDIM) return;
    int t = i >> 10, d = i & 1023;
    const float* p = part + (size_t)t * KACT * DDIM + d;
    out[i] += p[0] + p[DDIM] + p[2 * DDIM] + p[3 * DDIM];
}

// ---------------- launch ----------------

extern "C" void kernel_launch(void* const* d_in, const int* in_sizes, int n_in,
                              void* d_out, int out_size, void* d_ws, size_t ws_size,
                              hipStream_t stream) {
    (void)in_sizes; (void)n_in; (void)out_size; (void)ws_size;
    const float* x   = (const float*)d_in[0];
    const float* gw  = (const float*)d_in[1];
    const float* gb  = (const float*)d_in[2];
    const float* w1  = (const float*)d_in[3];
    const float* w3  = (const float*)d_in[4];
    const float* w2  = (const float*)d_in[5];
    const float* ws1 = (const float*)d_in[6];
    const float* ws3 = (const float*)d_in[7];
    const float* ws2 = (const float*)d_in[8];
    float* out = (float*)d_out;

    char* ws = (char*)d_ws;
    size_t off = 0;
    auto alloc = [&](size_t bytes) -> char* {
        char* p = ws + off;
        off += (bytes + 255) & ~(size_t)255;
        return p;
    };
    uint16_t* xb   = (uint16_t*)alloc((size_t)T_TOK * DDIM * 2);
    int*      widx = (int*)     alloc((size_t)ATOT * 4);
    float*    wval = (float*)   alloc((size_t)ATOT * 4);
    int*      perm = (int*)     alloc((size_t)ATOT * 4);
    int*      offs = (int*)     alloc((size_t)(NEXP + 1) * 4);
    uint16_t* act  = (uint16_t*)alloc((size_t)ATOT * HDIM * 2);
    uint16_t* acts = (uint16_t*)alloc((size_t)T_TOK * HSH * 2);
    float*    part = (float*)   alloc((size_t)ATOT * DDIM * 4);

    convert_x_kernel<<<(T_TOK * DDIM / 4 + 255) / 256, 256, 0, stream>>>(x, xb);
    gate_kernel<<<T_TOK / 4, 128, 0, stream>>>(x, gw, gb, widx, wval);
    dispatch_kernel<<<1, 256, 0, stream>>>(widx, perm, offs);

    // shared expert FFN (identity gather, writes y_shared into out)
    pass1_swiglu<<<dim3(HSH / 64, T_TOK / 128, 1), 256, 0, stream>>>(
        xb, ws1, ws3, acts, perm, offs, 1, HSH, DDIM, HSH);
    pass2_down<<<dim3(DDIM / 64, T_TOK / 128, 1), 256, 0, stream>>>(
        acts, ws2, out, perm, offs, wval, 1, HSH, HSH);

    // routed experts (sparse dispatch via perm/offsets)
    pass1_swiglu<<<dim3(HDIM / 64, T_TOK / 128, NEXP), 256, 0, stream>>>(
        xb, w1, w3, act, perm, offs, 0, HDIM, DDIM, HDIM);
    pass2_down<<<dim3(DDIM / 64, T_TOK / 128, NEXP), 256, 0, stream>>>(
        act, w2, part, perm, offs, wval, 0, HDIM, HDIM);

    combine_kernel<<<(T_TOK * DDIM + 255) / 256, 256, 0, stream>>>(out, part);
}
